// ExpertChoiceMoELayer_71047349010621
// MI455X (gfx1250) — matrix operand order repaired and checked by reading, not compiled, not run on hardware
//
#include <hip/hip_runtime.h>
#include <hip/hip_bf16.h>
#include <math.h>

typedef __attribute__((ext_vector_type(16))) _Float16 v16h;
typedef __attribute__((ext_vector_type(8)))  _Float16 v8h;
typedef __attribute__((ext_vector_type(4)))  _Float16 v4h;
typedef __attribute__((ext_vector_type(8)))  float    v8f;

#define H_DIM    1024
#define I_DIM    2816
#define N_EXP    8
#define N_TOK    16384
#define CAPACITY 2048
#define LN_EPS   1e-5f

// ---------------------------------------------------------------- reductions
__device__ __forceinline__ float block_reduce_sum(float v, float* sbuf) {
  const int tid = threadIdx.x;
  sbuf[tid] = v;
  __syncthreads();
  #pragma unroll
  for (int s = 128; s > 0; s >>= 1) {
    if (tid < s) sbuf[tid] += sbuf[tid + s];
    __syncthreads();
  }
  float r = sbuf[0];
  __syncthreads();
  return r;
}

__device__ __forceinline__ float block_reduce_max(float v, float* sbuf) {
  const int tid = threadIdx.x;
  sbuf[tid] = v;
  __syncthreads();
  #pragma unroll
  for (int s = 128; s > 0; s >>= 1) {
    if (tid < s) sbuf[tid] = fmaxf(sbuf[tid], sbuf[tid + s]);
    __syncthreads();
  }
  float r = sbuf[0];
  __syncthreads();
  return r;
}

__device__ __forceinline__ unsigned int order_key(float f) {
  unsigned int u = __float_as_uint(f);
  return (u & 0x80000000u) ? ~u : (u | 0x80000000u);
}

// ---------------------------------------------------------------- k0: zero
__global__ void k_zero(float* __restrict__ out, float* __restrict__ counts,
                       float* __restrict__ auxAcc) {
  size_t i = (size_t)blockIdx.x * blockDim.x + threadIdx.x;
  size_t total = (size_t)N_TOK * H_DIM;
  if (i < total) out[i] = 0.0f;
  if (i < N_TOK) counts[i] = 0.0f;
  if (i == 0) auxAcc[0] = 0.0f;
}

// ---------------------------------------------------------------- k1: LN + router
__global__ __launch_bounds__(256) void k_ln_router(
    const float* __restrict__ x, const float* __restrict__ gamma,
    const float* __restrict__ beta, const float* __restrict__ gw,
    float* __restrict__ logits, float* __restrict__ auxAcc) {
  __shared__ float sbuf[256];
  __shared__ float slog[N_EXP];
  const int t = blockIdx.x;
  const int tid = threadIdx.x;
  const float* xr = x + (size_t)t * H_DIM;

  float4 v = *(const float4*)(xr + tid * 4);
  float mean = block_reduce_sum(v.x + v.y + v.z + v.w, sbuf) * (1.0f / H_DIM);
  float sq = v.x * v.x + v.y * v.y + v.z * v.z + v.w * v.w;
  float var = block_reduce_sum(sq, sbuf) * (1.0f / H_DIM) - mean * mean;
  float rstd = rsqrtf(var + LN_EPS);

  float4 g = *(const float4*)(gamma + tid * 4);
  float4 b = *(const float4*)(beta + tid * 4);
  float xn0 = (v.x - mean) * rstd * g.x + b.x;
  float xn1 = (v.y - mean) * rstd * g.y + b.y;
  float xn2 = (v.z - mean) * rstd * g.z + b.z;
  float xn3 = (v.w - mean) * rstd * g.w + b.w;

  for (int e = 0; e < N_EXP; ++e) {
    float4 wv = *(const float4*)(gw + (size_t)e * H_DIM + tid * 4);
    float p = xn0 * wv.x + xn1 * wv.y + xn2 * wv.z + xn3 * wv.w;
    float tot = block_reduce_sum(p, sbuf);
    if (tid == 0) slog[e] = tot;
  }
  __syncthreads();
  if (tid < N_EXP) logits[(size_t)t * N_EXP + tid] = slog[tid];
  if (tid == 0) {
    float m = slog[0];
    #pragma unroll
    for (int e = 1; e < N_EXP; ++e) m = fmaxf(m, slog[e]);
    float se = 0.0f;
    #pragma unroll
    for (int e = 0; e < N_EXP; ++e) se += expf(slog[e] - m);
    float lse = m + logf(se);
    atomicAdd(auxAcc, lse * lse);
  }
}

// ---------------------------------------------------------------- k2: topk per expert
__global__ __launch_bounds__(256) void k_topk(
    const float* __restrict__ logits, int* __restrict__ top_idx,
    float* __restrict__ top_prob, float* __restrict__ counts) {
  const int e = blockIdx.x;
  const int tid = threadIdx.x;
  __shared__ float sbuf[256];
  __shared__ unsigned int hist[256];
  __shared__ unsigned int sPrefix;
  __shared__ int sNeed, sG;
  __shared__ float sMax, sInvDen;
  __shared__ int cntG, cntT;

  // token-softmax max + denominator for this expert column
  float m = -INFINITY;
  for (int t = tid; t < N_TOK; t += 256)
    m = fmaxf(m, logits[(size_t)t * N_EXP + e]);
  m = block_reduce_max(m, sbuf);
  if (tid == 0) sMax = m;
  __syncthreads();
  float sum = 0.0f;
  for (int t = tid; t < N_TOK; t += 256)
    sum += expf(logits[(size_t)t * N_EXP + e] - sMax);
  sum = block_reduce_sum(sum, sbuf);
  if (tid == 0) {
    sInvDen = 1.0f / sum;
    sPrefix = 0u; sNeed = CAPACITY; sG = 0;
  }
  __syncthreads();

  // 4-pass MSB radix select for the CAPACITY-th largest key
  for (int p = 3; p >= 0; --p) {
    hist[tid] = 0u;
    __syncthreads();
    unsigned int pref = sPrefix;
    for (int t = tid; t < N_TOK; t += 256) {
      unsigned int k = order_key(logits[(size_t)t * N_EXP + e]);
      bool match = (p == 3) || ((k >> ((p + 1) * 8)) == pref);
      if (match) atomicAdd(&hist[(k >> (p * 8)) & 255u], 1u);
    }
    __syncthreads();
    if (tid == 0) {
      unsigned int cum = 0; int sel = 0;
      for (int bkt = 255; bkt >= 0; --bkt) {
        if ((int)(cum + hist[bkt]) >= sNeed) { sel = bkt; break; }
        cum += hist[bkt];
      }
      sPrefix = (sPrefix << 8) | (unsigned int)sel;
      sNeed -= (int)cum;
      sG += (int)cum;
    }
    __syncthreads();
  }
  const unsigned int T = sPrefix;
  const int G = sG, R = sNeed;  // G strictly-greater, R ties needed
  if (tid == 0) { cntG = 0; cntT = 0; }
  __syncthreads();

  for (int t = tid; t < N_TOK; t += 256) {
    float l = logits[(size_t)t * N_EXP + e];
    unsigned int k = order_key(l);
    int slot = -1;
    if (k > T) slot = atomicAdd(&cntG, 1);
    else if (k == T) { int tp = atomicAdd(&cntT, 1); if (tp < R) slot = G + tp; }
    if (slot >= 0) {
      float prob = expf(l - sMax) * sInvDen;
      top_idx[(size_t)e * CAPACITY + slot] = t;
      top_prob[(size_t)e * CAPACITY + slot] = prob;
      atomicAdd(&counts[t], prob);
    }
  }
}

// ---------------------------------------------------------------- k3: gather + cvt f16
__global__ __launch_bounds__(256) void k_gather(
    const float* __restrict__ x, const int* __restrict__ top_idx,
    _Float16* __restrict__ xe) {
  const int ec = blockIdx.x;           // e*CAPACITY + c
  const int tid = threadIdx.x;
  const int row = top_idx[ec];
  float4 v = *(const float4*)(x + (size_t)row * H_DIM + tid * 4);
  v4h o;
  o[0] = (_Float16)v.x; o[1] = (_Float16)v.y;
  o[2] = (_Float16)v.z; o[3] = (_Float16)v.w;
  *(v4h*)(xe + (size_t)ec * H_DIM + tid * 4) = o;
}

// ---------------------------------------------------------------- WMMA fragment loads
// A: 16x32 f16, LDS tile row-major [16][32]
__device__ __forceinline__ v16h load_frag_a(const _Float16* s, int lane) {
  const _Float16* p = s + (lane & 15) * 32 + ((lane & 16) ? 8 : 0);
  v8h lo = *(const v8h*)p;
  v8h hi = *(const v8h*)(p + 16);
  v16h r;
  #pragma unroll
  for (int i = 0; i < 8; ++i) { r[i] = lo[i]; r[i + 8] = hi[i]; }
  return r;
}
// B: 32x16 f16, staged transposed in LDS as [n][k] (16 rows x 32 K)
__device__ __forceinline__ v16h load_frag_b(const _Float16* s, int lane) {
  const _Float16* p = s + (lane & 15) * 32 + ((lane & 16) ? 8 : 0);
  v8h lo = *(const v8h*)p;
  v8h hi = *(const v8h*)(p + 16);
  v16h r;
  #pragma unroll
  for (int i = 0; i < 8; ++i) { r[i] = lo[i]; r[i + 8] = hi[i]; }
  return r;
}

#define WMMA_F16(a, b, c) \
  __builtin_amdgcn_wmma_f32_16x16x32_f16(false, (a), false, (b), (short)0, (c), false, false)

// ---------------------------------------------------------------- gemm1: gate+up+SiLU
// block tile 128(M) x 128(N), 8 waves: wm=wave>>1 (4 over M), wn=wave&1 (2 over N)
// wave tile 32(M) x 64(N) for each of the two matrices (gate, up)
__global__ __launch_bounds__(256) void moe_gemm1(
    const _Float16* __restrict__ xe,   // [E*CAP, H]
    const float* __restrict__ wg,      // [E, I, H]
    const float* __restrict__ wu,      // [E, I, H]
    _Float16* __restrict__ hbuf) {     // [E*CAP, I]
  const int e  = blockIdx.z;
  const int m0 = blockIdx.y * 128;
  const int n0 = blockIdx.x * 128;
  const int tid = threadIdx.x;
  const int lane = tid & 31;
  const int wave = tid >> 5;
  const int wm = wave >> 1;      // 0..3 -> M offset 32*wm
  const int wn = wave & 1;       // 0..1 -> N offset 64*wn

  __shared__ __align__(16) _Float16 sA[128][32];
  __shared__ __align__(16) _Float16 sBg[128][32];
  __shared__ __align__(16) _Float16 sBu[128][32];

  const _Float16* xeB = xe + (size_t)(e * CAPACITY + m0) * H_DIM;
  const float* wgB = wg + (size_t)e * I_DIM * H_DIM + (size_t)n0 * H_DIM;
  const float* wuB = wu + (size_t)e * I_DIM * H_DIM + (size_t)n0 * H_DIM;

  v8f zero = {0.f, 0.f, 0.f, 0.f, 0.f, 0.f, 0.f, 0.f};
  v8f accG[2][4], accU[2][4];
  #pragma unroll
  for (int i = 0; i < 2; ++i)
    #pragma unroll
    for (int j = 0; j < 4; ++j) { accG[i][j] = zero; accU[i][j] = zero; }

  const int r128 = tid >> 1;            // 0..127 staging row
  const int cH   = (tid & 1) * 16;      // half-chunk (16 halves / 16 floats)

  for (int k0 = 0; k0 < H_DIM; k0 += 32) {
    __syncthreads();
    // stage A: 128 rows x 32 halves
    *(v16h*)&sA[r128][cH]     = *(const v16h*)(xeB + (size_t)r128 * H_DIM + k0 + cH);
    // stage Bg/Bu: 128 rows x 32 K, fp32 -> f16
    {
      const float* pg = wgB + (size_t)r128 * H_DIM + k0 + cH;
      const float* pu = wuB + (size_t)r128 * H_DIM + k0 + cH;
      __builtin_prefetch(pg + 32, 0, 1);   // next K tile -> global_prefetch_b8
      __builtin_prefetch(pu + 32, 0, 1);
      v8h g0, g1, u0, u1;
      #pragma unroll
      for (int i = 0; i < 8; ++i) {
        g0[i] = (_Float16)pg[i]; g1[i] = (_Float16)pg[i + 8];
        u0[i] = (_Float16)pu[i]; u1[i] = (_Float16)pu[i + 8];
      }
      *(v8h*)&sBg[r128][cH] = g0; *(v8h*)&sBg[r128][cH + 8] = g1;
      *(v8h*)&sBu[r128][cH] = u0; *(v8h*)&sBu[r128][cH + 8] = u1;
    }
    __syncthreads();

    v16h aF[2];
    aF[0] = load_frag_a(&sA[wm * 32][0], lane);
    aF[1] = load_frag_a(&sA[wm * 32 + 16][0], lane);
    #pragma unroll
    for (int nt = 0; nt < 4; ++nt) {
      v16h gB = load_frag_b(&sBg[wn * 64 + nt * 16][0], lane);
      v16h uB = load_frag_b(&sBu[wn * 64 + nt * 16][0], lane);
      #pragma unroll
      for (int mt = 0; mt < 2; ++mt) {
        accG[mt][nt] = WMMA_F16(aF[mt], gB, accG[mt][nt]);
        accU[mt][nt] = WMMA_F16(aF[mt], uB, accU[mt][nt]);
      }
    }
  }

  // epilogue: h = silu(g) * u  -> f16
  const int mlo = (lane & 16) ? 8 : 0;
  const int nlo = lane & 15;
  #pragma unroll
  for (int mt = 0; mt < 2; ++mt)
    #pragma unroll
    for (int nt = 0; nt < 4; ++nt)
      #pragma unroll
      for (int r = 0; r < 8; ++r) {
        int mrow = m0 + wm * 32 + mt * 16 + r + mlo;
        int ncol = n0 + wn * 64 + nt * 16 + nlo;
        float gv = accG[mt][nt][r];
        float uv = accU[mt][nt][r];
        float hv = (gv / (1.0f + expf(-gv))) * uv;
        hbuf[(size_t)(e * CAPACITY + mrow) * I_DIM + ncol] = (_Float16)hv;
      }
}

// ---------------------------------------------------------------- gemm2: down-proj
// block tile 128(M) x 128(N), wave tile 32x64
__global__ __launch_bounds__(256) void moe_gemm2(
    const _Float16* __restrict__ hbuf,   // [E*CAP, I]
    const float* __restrict__ wd,        // [E, H, I]
    const int* __restrict__ top_idx,
    const float* __restrict__ top_prob,
    float* __restrict__ out) {
  const int e  = blockIdx.z;
  const int m0 = blockIdx.y * 128;
  const int n0 = blockIdx.x * 128;
  const int tid = threadIdx.x;
  const int lane = tid & 31;
  const int wave = tid >> 5;
  const int wm = wave >> 1;
  const int wn = wave & 1;

  __shared__ __align__(16) _Float16 sA[128][32];
  __shared__ __align__(16) _Float16 sB[128][32];

  const _Float16* hB = hbuf + (size_t)(e * CAPACITY + m0) * I_DIM;
  const float* wdB = wd + (size_t)e * H_DIM * I_DIM + (size_t)n0 * I_DIM;

  v8f zero = {0.f, 0.f, 0.f, 0.f, 0.f, 0.f, 0.f, 0.f};
  v8f acc[2][4];
  #pragma unroll
  for (int i = 0; i < 2; ++i)
    #pragma unroll
    for (int j = 0; j < 4; ++j) acc[i][j] = zero;

  const int r128 = tid >> 1;
  const int cH   = (tid & 1) * 16;

  for (int k0 = 0; k0 < I_DIM; k0 += 32) {
    __syncthreads();
    *(v16h*)&sA[r128][cH] = *(const v16h*)(hB + (size_t)r128 * I_DIM + k0 + cH);
    {
      const float* pb = wdB + (size_t)r128 * I_DIM + k0 + cH;
      __builtin_prefetch(pb + 32, 0, 1);
      v8h b0, b1;
      #pragma unroll
      for (int i = 0; i < 8; ++i) { b0[i] = (_Float16)pb[i]; b1[i] = (_Float16)pb[i + 8]; }
      *(v8h*)&sB[r128][cH] = b0; *(v8h*)&sB[r128][cH + 8] = b1;
    }
    __syncthreads();

    v16h aF[2];
    aF[0] = load_frag_a(&sA[wm * 32][0], lane);
    aF[1] = load_frag_a(&sA[wm * 32 + 16][0], lane);
    #pragma unroll
    for (int nt = 0; nt < 4; ++nt) {
      v16h bF = load_frag_b(&sB[wn * 64 + nt * 16][0], lane);
      #pragma unroll
      for (int mt = 0; mt < 2; ++mt)
        acc[mt][nt] = WMMA_F16(aF[mt], bF, acc[mt][nt]);
    }
  }

  const int mlo = (lane & 16) ? 8 : 0;
  const int nlo = lane & 15;
  #pragma unroll
  for (int mt = 0; mt < 2; ++mt) {
    #pragma unroll
    for (int r = 0; r < 8; ++r) {
      int mrow = m0 + wm * 32 + mt * 16 + r + mlo;
      int tok = top_idx[(size_t)e * CAPACITY + mrow];
      float p = top_prob[(size_t)e * CAPACITY + mrow];
      #pragma unroll
      for (int nt = 0; nt < 4; ++nt) {
        int ncol = n0 + wn * 64 + nt * 16 + nlo;
        atomicAdd(&out[(size_t)tok * H_DIM + ncol], p * acc[mt][nt][r]);
      }
    }
  }
}

// ---------------------------------------------------------------- k6: normalize + aux
__global__ void k_norm(float* __restrict__ out, const float* __restrict__ counts,
                       const float* __restrict__ auxAcc) {
  size_t i = (size_t)blockIdx.x * blockDim.x + threadIdx.x;
  size_t total = (size_t)N_TOK * H_DIM;
  if (i < total) {
    int t = (int)(i >> 10);  // H_DIM == 1024
    out[i] = out[i] / fmaxf(counts[t], 1e-5f);
  }
  if (i == 0) out[total] = auxAcc[0] * (0.001f / (float)N_TOK);
}

// ---------------------------------------------------------------- launch
extern "C" void kernel_launch(void* const* d_in, const int* in_sizes, int n_in,
                              void* d_out, int out_size, void* d_ws, size_t ws_size,
                              hipStream_t stream) {
  const float* x     = (const float*)d_in[0];
  const float* gamma = (const float*)d_in[1];
  const float* beta  = (const float*)d_in[2];
  const float* gw    = (const float*)d_in[3];
  const float* wg    = (const float*)d_in[4];
  const float* wu    = (const float*)d_in[5];
  const float* wd    = (const float*)d_in[6];
  float* out = (float*)d_out;

  char* w = (char*)d_ws;
  auto carve = [&](size_t bytes) -> char* {
    char* p = w;
    w += (bytes + 255) & ~(size_t)255;
    return p;
  };
  float*    logits   = (float*)carve((size_t)N_TOK * N_EXP * sizeof(float));
  float*    counts   = (float*)carve((size_t)N_TOK * sizeof(float));
  float*    auxAcc   = (float*)carve(256);
  int*      top_idx  = (int*)carve((size_t)N_EXP * CAPACITY * sizeof(int));
  float*    top_prob = (float*)carve((size_t)N_EXP * CAPACITY * sizeof(float));
  _Float16* xe       = (_Float16*)carve((size_t)N_EXP * CAPACITY * H_DIM * sizeof(_Float16));
  _Float16* hbuf     = (_Float16*)carve((size_t)N_EXP * CAPACITY * I_DIM * sizeof(_Float16));

  const size_t total = (size_t)N_TOK * H_DIM;
  k_zero<<<(unsigned)((total + 255) / 256), 256, 0, stream>>>(out, counts, auxAcc);
  k_ln_router<<<N_TOK, 256, 0, stream>>>(x, gamma, beta, gw, logits, auxAcc);
  k_topk<<<N_EXP, 256, 0, stream>>>(logits, top_idx, top_prob, counts);
  k_gather<<<N_EXP * CAPACITY, 256, 0, stream>>>(x, top_idx, xe);

  dim3 g1(I_DIM / 128, CAPACITY / 128, N_EXP);
  moe_gemm1<<<g1, 256, 0, stream>>>(xe, wg, wu, hbuf);

  dim3 g2(H_DIM / 128, CAPACITY / 128, N_EXP);
  moe_gemm2<<<g2, 256, 0, stream>>>(hbuf, wd, top_idx, top_prob, out);

  k_norm<<<(unsigned)((total + 255) / 256), 256, 0, stream>>>(out, counts, auxAcc);
}
